// GRU_24163486007466
// MI455X (gfx1250) — matrix operand order repaired and checked by reading, not compiled, not run on hardware
//
#include <hip/hip_runtime.h>
#include <hip/hip_bf16.h>

// ---------------------------------------------------------------------------
// GRU stack for MI455X (gfx1250), wave32 + WMMA.
//   P1/P2/P3: convert/pack weights+embeddings to bf16 in WMMA fragment layout
//   gemm_gx : gx = emb @ w_ih^T + b_ih via v_wmma_f32_16x16x32_bf16
//   gru1    : 16384-step recurrence, 16 cooperating WGs; 304KB of the per-WG
//             weight slice cached in CDNA5's 320KB LDS (rest streams via
//             WGP$/L2); cluster barrier + atomic fallback barrier per step
//   gru2_fc : tiny scalar GRU + final FC
// Workspace requirement: ~222 MB.
// ---------------------------------------------------------------------------

typedef __attribute__((ext_vector_type(16))) __bf16 v16bf;
typedef __attribute__((ext_vector_type(8)))  float  v8f;

#define T1      16384   // NUM_SEQ * CHUNK
#define EDIM    256
#define H1      1024
#define G3      3072    // 3 * H1
#define CHUNK_  1024
#define NWG     16      // workgroups cooperating on the recurrence
#define NCROW   152     // weight rows cached in LDS per WG (of 192), 304 KB

__device__ __forceinline__ unsigned f2bf(float f) {
    unsigned u = __float_as_uint(f);
    return (u + 0x7FFFu + ((u >> 16) & 1u)) >> 16;   // RNE
}
__device__ __forceinline__ float bf2f(unsigned s) {
    return __uint_as_float(s << 16);
}
__device__ __forceinline__ float sigm(float x) {
    return 1.0f / (1.0f + __expf(-x));
}

// --- P1: w_hh -> bf16 row-major (packed dword pairs), also zero barrier ctr
__global__ void prep_whh(const float* __restrict__ w_hh, unsigned* __restrict__ whh,
                         int* __restrict__ ctr) {
    int idx = blockIdx.x * 256 + threadIdx.x;
    if (idx == 0) *ctr = 0;
    if (idx < G3 * (H1 / 2)) {
        int row = idx >> 9;            // /512
        int k   = (idx & 511) * 2;
        unsigned lo = f2bf(w_hh[row * H1 + k]);
        unsigned hi = f2bf(w_hh[row * H1 + k + 1]);
        whh[idx] = lo | (hi << 16);
    }
}

// --- P2: w_ih -> bf16 B-fragment layout.
// B tile (kb,nt): dword p of lane L = pair (k = kb*32 + (p>>2)*16 + (L>>4)*8 + (p&3)*2, n = nt*16 + (L&15))
__global__ void prep_bsw(const float* __restrict__ w_ih, unsigned* __restrict__ Bsw) {
    int f = blockIdx.x * 256 + threadIdx.x;
    if (f >= (G3 / 16) * (EDIM / 32) * 32 * 8) return;   // 393216
    int p  = f & 7;
    int L  = (f >> 3) & 31;
    int ti = f >> 8;
    int kb = ti & 7;
    int nt = ti >> 3;
    int n = nt * 16 + (L & 15);
    int k = kb * 32 + ((p >> 2) * 16) + ((L >> 4) * 8) + ((p & 3) * 2);
    unsigned lo = f2bf(w_ih[n * EDIM + k]);
    unsigned hi = f2bf(w_ih[n * EDIM + k + 1]);
    Bsw[f] = lo | (hi << 16);
}

// --- P3: embedding gather -> bf16 A-fragment layout.
// A tile (mt,kb): element e of lane L = (M = mt*16 + (L&15),
//                 K = kb*32 + (e>>3)*16 + (L>>4)*8 + (e&7))
__global__ void prep_asw(const int* __restrict__ x, const float* __restrict__ E,
                         unsigned* __restrict__ Asw) {
    int f = blockIdx.x * 256 + threadIdx.x;
    if (f >= T1 * EDIM / 2) return;                      // 2097152
    int p  = f & 7;
    int L  = (f >> 3) & 31;
    int kb = (f >> 8) & 7;
    int mt = f >> 11;
    int row = mt * 16 + (L & 15);
    int K   = kb * 32 + ((p >> 2) * 16) + ((L >> 4) * 8) + ((p & 3) * 2);
    int tok = x[row];
    unsigned lo = f2bf(E[tok * EDIM + K]);
    unsigned hi = f2bf(E[tok * EDIM + K + 1]);
    Asw[f] = lo | (hi << 16);
}

// --- GEMM: gx[T1][G3] = A[T1][EDIM] * B[EDIM][G3] + b_ih, bf16 WMMA, f32 acc.
// Block = 128x128 tile, 8 waves, wave = 64x32 (4x2 WMMA tiles), K-loop 8x32.
__global__ __launch_bounds__(256) void gemm_gx(const unsigned short* __restrict__ Asw,
                                               const unsigned short* __restrict__ Bsw,
                                               const float* __restrict__ b_ih,
                                               float* __restrict__ gx) {
    const int lane = threadIdx.x & 31;
    const int w    = threadIdx.x >> 5;
    const int m0 = blockIdx.x * 128 + (w & 1) * 64;
    const int n0 = blockIdx.y * 128 + (w >> 1) * 32;

    v8f acc[4][2] = {};

#pragma unroll
    for (int kb = 0; kb < 8; ++kb) {
        v16bf a[4], b[2];
#pragma unroll
        for (int i = 0; i < 4; ++i) {
            int mt = (m0 >> 4) + i;
            a[i] = *reinterpret_cast<const v16bf*>(Asw + ((mt * 8 + kb) * 32 + lane) * 16);
        }
#pragma unroll
        for (int j = 0; j < 2; ++j) {
            int nt = (n0 >> 4) + j;
            b[j] = *reinterpret_cast<const v16bf*>(Bsw + (nt * 8 + kb) * 512 + lane * 16);
        }
#pragma unroll
        for (int i = 0; i < 4; ++i)
#pragma unroll
            for (int j = 0; j < 2; ++j)
                acc[i][j] = __builtin_amdgcn_wmma_f32_16x16x32_bf16(
                    false, a[i], false, b[j], (short)0, acc[i][j], false, false);
    }

#pragma unroll
    for (int j = 0; j < 2; ++j) {
        int col = n0 + j * 16 + (lane & 15);
        float bias = b_ih[col];
#pragma unroll
        for (int i = 0; i < 4; ++i) {
            int rbase = m0 + i * 16 + ((lane >> 4) * 8);
#pragma unroll
            for (int v = 0; v < 8; ++v)
                gx[(size_t)(rbase + v) * G3 + col] = acc[i][j][v] + bias;
        }
    }
}

// --- Recurrence: 16 WGs x 1024 threads. WG w owns hidden units [w*64, w*64+64),
// computing all three gate rows for them (192 mat-vec columns of length 1024).
// First NCROW rows live in LDS (loaded once); the rest stream via WGP$/L2.
__global__ __launch_bounds__(1024) void gru1_kernel(const unsigned* __restrict__ whh,
                                                    const float* __restrict__ gx,
                                                    const float* __restrict__ b_hh,
                                                    float* hglob,
                                                    float* __restrict__ hs_last,
                                                    int* ctr) {
    __shared__ unsigned lds_w[NCROW * 512];   // 304 KB bf16x2 weight cache
    __shared__ float    lds_h[H1];            //   4 KB
    __shared__ float    lds_a[192];
    const int tid  = threadIdx.x;
    const int lane = tid & 31;
    const int wv   = tid >> 5;
    const int wg   = blockIdx.x;

    // one-time cooperative preload of the hot weight rows into LDS
    for (int d = tid; d < NCROW * 512; d += 1024) {
        int lc  = d >> 9;
        int row = (lc >> 6) * H1 + wg * 64 + (lc & 63);
        lds_w[d] = whh[row * 512 + (d & 511)];
    }
    __syncthreads();

    for (int t = 0; t < T1; ++t) {
        // fresh h (device-scope load: bypass stale WGP$)
        lds_h[tid] = (t == 0) ? 0.0f
            : __hip_atomic_load(&hglob[tid], __ATOMIC_RELAXED, __HIP_MEMORY_SCOPE_AGENT);
        __syncthreads();

        // 32 waves x 6 columns each; lanes along K, bf16x2 weight dwords
#pragma unroll
        for (int j = 0; j < 6; ++j) {
            int lc  = wv * 6 + j;                         // 0..191
            float acc = 0.0f;
            if (lc < NCROW) {
                const unsigned* wr = &lds_w[lc * 512];
#pragma unroll
                for (int i = 0; i < 16; ++i) {
                    unsigned u = wr[i * 32 + lane];
                    float2 hp = reinterpret_cast<const float2*>(lds_h)[i * 32 + lane];
                    acc += bf2f(u & 0xFFFFu) * hp.x + bf2f(u >> 16) * hp.y;
                }
            } else {
                int row = (lc >> 6) * H1 + wg * 64 + (lc & 63);
                const unsigned* wr = whh + row * 512;
#pragma unroll
                for (int i = 0; i < 16; ++i) {
                    unsigned u = wr[i * 32 + lane];
                    float2 hp = reinterpret_cast<const float2*>(lds_h)[i * 32 + lane];
                    acc += bf2f(u & 0xFFFFu) * hp.x + bf2f(u >> 16) * hp.y;
                }
            }
            acc += __shfl_xor(acc, 16, 32);
            acc += __shfl_xor(acc, 8, 32);
            acc += __shfl_xor(acc, 4, 32);
            acc += __shfl_xor(acc, 2, 32);
            acc += __shfl_xor(acc, 1, 32);
            if (lane == 0) lds_a[lc] = acc;
        }
        __syncthreads();

        if (tid < 64) {
            int jg = wg * 64 + tid;
            const float* gxt = gx + (size_t)t * G3;
            float r = sigm(gxt[jg]        + lds_a[tid]        + b_hh[jg]);
            float z = sigm(gxt[H1 + jg]   + lds_a[64 + tid]   + b_hh[H1 + jg]);
            float n = tanhf(gxt[2*H1 + jg] + r * (lds_a[128 + tid] + b_hh[2*H1 + jg]));
            float hnew = (1.0f - z) * n + z * lds_h[jg];
            __hip_atomic_store(&hglob[jg], hnew, __ATOMIC_RELAXED, __HIP_MEMORY_SCOPE_AGENT);
            if (t >= T1 - CHUNK_)
                hs_last[(size_t)(t - (T1 - CHUNK_)) * H1 + jg] = hnew;
        }
        __threadfence();
        __syncthreads();

        // CDNA5 cluster barrier (s_barrier_signal/-wait -3): fast path when
        // dispatched as a 16-WG cluster; architectural NOP otherwise.
        __builtin_amdgcn_s_cluster_barrier();

        // Portable fallback: monotonic agent-scope counter barrier.
        if (wv == 0 && lane == 0) {
            atomicAdd(ctr, 1);
            int target = (t + 1) * NWG;
            while (__hip_atomic_load(ctr, __ATOMIC_RELAXED, __HIP_MEMORY_SCOPE_AGENT) < target)
                __builtin_amdgcn_s_sleep(1);
        }
        __syncthreads();
    }
}

// --- Layer-2 GRU (H=1) + final FC, single workgroup.
__global__ __launch_bounds__(1024) void gru2_fc_kernel(const float* __restrict__ hs_last,
                                                       const float* __restrict__ w_ih2,
                                                       const float* __restrict__ w_hh2,
                                                       const float* __restrict__ b_ih2,
                                                       const float* __restrict__ b_hh2,
                                                       const float* __restrict__ fc2_w,
                                                       const float* __restrict__ fc2_b,
                                                       float* __restrict__ out) {
    __shared__ float g2r[CHUNK_], g2z[CHUNK_], g2n[CHUNK_], sq[CHUNK_];
    const int tid = threadIdx.x;

    const float* hrow = hs_last + (size_t)tid * H1;
    float a0 = 0.f, a1 = 0.f, a2 = 0.f;
    for (int k = 0; k < H1; ++k) {
        float h = hrow[k];
        a0 += h * w_ih2[k];
        a1 += h * w_ih2[H1 + k];
        a2 += h * w_ih2[2 * H1 + k];
    }
    g2r[tid] = a0 + b_ih2[0];
    g2z[tid] = a1 + b_ih2[1];
    g2n[tid] = a2 + b_ih2[2];
    __syncthreads();

    if (tid == 0) {
        float h = 0.f;
        float w0 = w_hh2[0], w1 = w_hh2[1], w2 = w_hh2[2];
        float c0 = b_hh2[0], c1 = b_hh2[1], c2 = b_hh2[2];
        for (int t = 0; t < CHUNK_; ++t) {
            float r = sigm(g2r[t] + h * w0 + c0);
            float z = sigm(g2z[t] + h * w1 + c1);
            float n = tanhf(g2n[t] + r * (h * w2 + c2));
            h = (1.0f - z) * n + z * h;
            sq[t] = h;
        }
    }
    __syncthreads();

    float v = sq[tid];
    g2r[tid] = v * fc2_w[tid];
    g2z[tid] = v * fc2_w[CHUNK_ + tid];
    __syncthreads();
    for (int s = 512; s > 0; s >>= 1) {
        if (tid < s) { g2r[tid] += g2r[tid + s]; g2z[tid] += g2z[tid + s]; }
        __syncthreads();
    }
    if (tid == 0) {
        out[0] = g2r[0] + fc2_b[0];
        out[1] = g2z[0] + fc2_b[1];
    }
}

extern "C" void kernel_launch(void* const* d_in, const int* in_sizes, int n_in,
                              void* d_out, int out_size, void* d_ws, size_t ws_size,
                              hipStream_t stream) {
    (void)in_sizes; (void)n_in; (void)out_size; (void)ws_size;

    const int*   x      = (const int*)  d_in[0];
    const float* E      = (const float*)d_in[1];
    const float* w_ih   = (const float*)d_in[2];
    const float* w_hh   = (const float*)d_in[3];
    const float* b_ih   = (const float*)d_in[4];
    const float* b_hh   = (const float*)d_in[5];
    const float* w_ih2  = (const float*)d_in[6];
    const float* w_hh2  = (const float*)d_in[7];
    const float* b_ih2  = (const float*)d_in[8];
    const float* b_hh2  = (const float*)d_in[9];
    const float* fc2_w  = (const float*)d_in[10];
    const float* fc2_b  = (const float*)d_in[11];
    float* out = (float*)d_out;

    // workspace layout (bytes)
    char* ws = (char*)d_ws;
    unsigned* Asw     = (unsigned*)(ws);                          //   8,388,608
    unsigned* Bsw     = (unsigned*)(ws + 8388608);                //   1,572,864
    unsigned* whh     = (unsigned*)(ws + 9961472);                //   6,291,456
    float*    gx      = (float*)   (ws + 16252928);               // 201,326,592
    float*    hglob   = (float*)   (ws + 217579520);              //       4,096
    float*    hs_last = (float*)   (ws + 217583616);              //   4,194,304
    int*      ctr     = (int*)     (ws + 221777920);              // total ~222MB

    prep_whh<<<6144, 256, 0, stream>>>(w_hh, whh, ctr);
    prep_bsw<<<1536, 256, 0, stream>>>(w_ih, Bsw);
    prep_asw<<<8192, 256, 0, stream>>>(x, E, Asw);

    dim3 ggrid(T1 / 128, G3 / 128);   // 128 x 24
    gemm_gx<<<ggrid, 256, 0, stream>>>((const unsigned short*)Asw,
                                       (const unsigned short*)Bsw, b_ih, gx);

    gru1_kernel<<<NWG, 1024, 0, stream>>>(whh, gx, b_hh, hglob, hs_last, ctr);

    gru2_fc_kernel<<<1, 1024, 0, stream>>>(hs_last, w_ih2, w_hh2, b_ih2, b_hh2,
                                           fc2_w, fc2_b, out);
}
